// MultiHeadAttention_31001073942591
// MI455X (gfx1250) — matrix operand order repaired and checked by reading, not compiled, not run on hardware
//
#include <hip/hip_runtime.h>
#include <hip/hip_bf16.h>

// ---------------------------------------------------------------------------
// Multi-head attention, MI455X (gfx1250, wave32, WMMA f16 -> f32 accum).
//   B=4, S=2048, D_MODEL=1024, H=16, D_K=64
// All four GEMM stages run on v_wmma_f32_16x16x32_f16.
// ---------------------------------------------------------------------------

#define BATCH   4
#define SEQ     2048
#define DM      1024
#define NHEAD   16
#define DK      64
#define MROWS   (BATCH * SEQ)   // 8192

typedef __attribute__((ext_vector_type(16))) _Float16 v16h;
typedef __attribute__((ext_vector_type(8)))  _Float16 v8h;
typedef __attribute__((ext_vector_type(8)))  float    v8f;

// D = A(16x32 f16) * B(32x16 f16) + C(16x16 f32)
__device__ __forceinline__ v8f wmma_f16(v16h a, v16h b, v8f c) {
  return __builtin_amdgcn_wmma_f32_16x16x32_f16(
      /*neg_a=*/false, a, /*neg_b=*/false, b,
      /*c_mod=*/(short)0, c, /*reuse_a=*/false, /*reuse_b=*/false);
}

// A-fragment (16x32, 16-bit): lane L holds row M = L%16.
//   halves 0..7  : K = 8*(L/16) + 0..7        (contiguous, 16B)
//   halves 8..15 : K = 16 + 8*(L/16) + 0..7   (contiguous, 16B at p+16)
// `p` must already point at element (row, k0 + 8*(L/16)).
__device__ __forceinline__ v16h load_a_f16(const _Float16* p) {
  v8h lo = *(const v8h*)(p);
  v8h hi = *(const v8h*)(p + 16);
  v16h r;
#pragma unroll
  for (int i = 0; i < 8; ++i) { r[i] = lo[i]; r[i + 8] = hi[i]; }
  return r;
}

// Same fragment, loaded from an f32 source with in-register f16 conversion.
__device__ __forceinline__ v16h load_a_f32(const float* p) {
  v16h r;
#pragma unroll
  for (int i = 0; i < 8; ++i) {
    r[i]     = (_Float16)p[i];
    r[i + 8] = (_Float16)p[i + 16];
  }
  return r;
}

// B-fragment (32x16, 16-bit): lane L holds column N = L%16,
// halves 0..7 = K = 8*(L/16) + 0..7, halves 8..15 = K = 16 + 8*(L/16) + 0..7.
// With a K-major (transposed) source these are two 16-byte loads per lane.
__device__ __forceinline__ v16h load_b_f16(const _Float16* p) {
  v8h lo = *(const v8h*)(p);
  v8h hi = *(const v8h*)(p + 16);
  v16h r;
#pragma unroll
  for (int i = 0; i < 8; ++i) { r[i] = lo[i]; r[i + 8] = hi[i]; }
  return r;
}

// ---------------------------------------------------------------------------
// Kernel 0: W[k][n] (f32, row-major [1024x1024]) -> Wt[n][k] (f16)
// ---------------------------------------------------------------------------
__global__ __launch_bounds__(256)
void wtrans_kernel(const float* __restrict__ w, _Float16* __restrict__ wt) {
  __shared__ _Float16 tile[32][33];
  const int tx = threadIdx.x, ty = threadIdx.y;
  const int n0 = blockIdx.x * 32, k0 = blockIdx.y * 32;
#pragma unroll
  for (int i = 0; i < 32; i += 8)
    tile[ty + i][tx] = (_Float16)w[(size_t)(k0 + ty + i) * DM + n0 + tx];
  __syncthreads();
#pragma unroll
  for (int i = 0; i < 32; i += 8)
    wt[(size_t)(n0 + ty + i) * DM + k0 + tx] = tile[tx][ty + i];
}

// ---------------------------------------------------------------------------
// Kernel 1: fused QKV projection.
//   Y = X @ W + b ;  X:[8192,1024] f32, Wt:[1024,1024] f16 (transposed)
//   z=0 -> Qh[B,H,S,DK] f16 ; z=1 -> Kh[B,H,S,DK] f16 ; z=2 -> Vt[B,H,DK,S] f16
// block = 128 threads (4 waves); wave tile 32x64; workgroup tile 128x64.
// ---------------------------------------------------------------------------
__global__ __launch_bounds__(128)
void qkv_proj_kernel(const float* __restrict__ qx, const float* __restrict__ kx,
                     const float* __restrict__ vx,
                     const _Float16* __restrict__ wtq,
                     const _Float16* __restrict__ wtk,
                     const _Float16* __restrict__ wtv,
                     const float* __restrict__ bq, const float* __restrict__ bk,
                     const float* __restrict__ bv,
                     _Float16* __restrict__ qh, _Float16* __restrict__ kh,
                     _Float16* __restrict__ vt) {
  const int z = blockIdx.z;
  const float*    x    = (z == 0) ? qx  : (z == 1) ? kx  : vx;
  const _Float16* wt   = (z == 0) ? wtq : (z == 1) ? wtk : wtv;
  const float*    bias = (z == 0) ? bq  : (z == 1) ? bk  : bv;

  const int lane = threadIdx.x & 31;
  const int wave = threadIdx.x >> 5;
  const int g    = lane >> 4;
  const int ln   = lane & 15;
  const int row0 = blockIdx.x * 128 + wave * 32;
  const int col0 = blockIdx.y * 64;

  v8f acc[2][4];
#pragma unroll
  for (int i = 0; i < 2; ++i)
#pragma unroll
    for (int j = 0; j < 4; ++j) acc[i][j] = (v8f)(0.0f);

  for (int kb = 0; kb < DM; kb += 32) {
    v16h a[2];
#pragma unroll
    for (int i = 0; i < 2; ++i)
      a[i] = load_a_f32(x + (size_t)(row0 + i * 16 + ln) * DM + kb + 8 * g);
#pragma unroll
    for (int j = 0; j < 4; ++j) {
      v16h bf = load_b_f16(wt + (size_t)(col0 + j * 16 + ln) * DM + kb + 8 * g);
#pragma unroll
      for (int i = 0; i < 2; ++i) acc[i][j] = wmma_f16(a[i], bf, acc[i][j]);
    }
  }

  // Epilogue: bias add, f16 convert, scatter to head-major layouts.
#pragma unroll
  for (int j = 0; j < 4; ++j) {
    const int n  = col0 + j * 16 + ln;
    const float bval = bias[n];
    const int hh = n >> 6;          // head
    const int d  = n & 63;          // feature within head
#pragma unroll
    for (int i = 0; i < 2; ++i) {
#pragma unroll
      for (int r = 0; r < 8; ++r) {
        const int m  = row0 + i * 16 + r + 8 * g;
        const int bb = m >> 11;     // batch
        const int s  = m & 2047;    // sequence position
        const _Float16 hv = (_Float16)(acc[i][j][r] + bval);
        if (z == 2) {
          vt[(((size_t)(bb * NHEAD + hh) * DK + d) << 11) + s] = hv;
        } else {
          _Float16* dst = (z == 0) ? qh : kh;
          dst[(((size_t)(bb * NHEAD + hh) << 11) + s) * DK + d] = hv;
        }
      }
    }
  }
}

// ---------------------------------------------------------------------------
// Kernel 2: flash attention over one (b,h), 128 query rows per workgroup.
// 4 waves x 32 query rows; stream 32-key blocks with online softmax.
// P (probabilities) are restaged C-layout -> A-layout through per-wave LDS.
// Output: Xh[B,S,DM] f16 (heads re-concatenated).
// ---------------------------------------------------------------------------
__global__ __launch_bounds__(128)
void attn_kernel(const _Float16* __restrict__ qh,
                 const _Float16* __restrict__ kh,
                 const _Float16* __restrict__ vt,
                 _Float16* __restrict__ xh) {
  __shared__ _Float16 ptile[4][2][16 * 40];   // [wave][rowtile][16 rows x 32 cols + pad]

  const int lane = threadIdx.x & 31;
  const int wave = threadIdx.x >> 5;
  const int g    = lane >> 4;
  const int ln   = lane & 15;
  const int bh   = blockIdx.y;                 // b*16 + h
  const int b    = bh >> 4;
  const int h    = bh & 15;
  const int qrow0 = blockIdx.x * 128 + wave * 32;

  const _Float16* qp = qh + (size_t)bh * SEQ * DK;
  const _Float16* kp = kh + (size_t)bh * SEQ * DK;
  const _Float16* vp = vt + (size_t)bh * DK * SEQ;

  // Q fragments for this wave's 32 rows, pre-scaled by 1/sqrt(DK)=0.125.
  v16h aq[2][2];
#pragma unroll
  for (int i = 0; i < 2; ++i)
#pragma unroll
    for (int kc = 0; kc < 2; ++kc) {
      aq[i][kc] =
          load_a_f16(qp + (size_t)(qrow0 + i * 16 + ln) * DK + kc * 32 + 8 * g);
#pragma unroll
      for (int e = 0; e < 16; ++e) aq[i][kc][e] *= (_Float16)0.125f;
    }

  float mrow[2][8], lrow[2][8];
  v8f o[2][4];
#pragma unroll
  for (int i = 0; i < 2; ++i) {
#pragma unroll
    for (int r = 0; r < 8; ++r) { mrow[i][r] = -3.0e38f; lrow[i][r] = 0.0f; }
#pragma unroll
    for (int ct = 0; ct < 4; ++ct) o[i][ct] = (v8f)(0.0f);
  }

  for (int kb = 0; kb < SEQ; kb += 32) {
    // ---- scores S = (Q/sqrt(dk)) @ K^T for 32 keys ----
    v8f sc[2][2];
#pragma unroll
    for (int i = 0; i < 2; ++i)
#pragma unroll
      for (int j = 0; j < 2; ++j) sc[i][j] = (v8f)(0.0f);

#pragma unroll
    for (int j = 0; j < 2; ++j)
#pragma unroll
      for (int kc = 0; kc < 2; ++kc) {
        v16h bk =
            load_b_f16(kp + (size_t)(kb + j * 16 + ln) * DK + kc * 32 + 8 * g);
#pragma unroll
        for (int i = 0; i < 2; ++i) sc[i][j] = wmma_f16(aq[i][kc], bk, sc[i][j]);
      }

    // ---- online softmax (rows live in (VGPR r, lane group g); cols in lanes) ----
#pragma unroll
    for (int i = 0; i < 2; ++i) {
      float al[8];
#pragma unroll
      for (int r = 0; r < 8; ++r) {
        float bm = fmaxf(sc[i][0][r], sc[i][1][r]);
#pragma unroll
        for (int off = 8; off >= 1; off >>= 1)
          bm = fmaxf(bm, __shfl_xor(bm, off, 32));
        const float mo = mrow[i][r];
        const float mn = fmaxf(mo, bm);
        const float alpha = __expf(mo - mn);
        const float p0 = __expf(sc[i][0][r] - mn);
        const float p1 = __expf(sc[i][1][r] - mn);
        sc[i][0][r] = p0;
        sc[i][1][r] = p1;
        float rs = p0 + p1;
#pragma unroll
        for (int off = 8; off >= 1; off >>= 1) rs += __shfl_xor(rs, off, 32);
        lrow[i][r] = lrow[i][r] * alpha + rs;
        mrow[i][r] = mn;
        al[r] = alpha;
      }
#pragma unroll
      for (int ct = 0; ct < 4; ++ct)
#pragma unroll
        for (int r = 0; r < 8; ++r) o[i][ct][r] *= al[r];

      // stage P as f16 in LDS (C layout -> row-major 16x32 tile)
      _Float16* pt = &ptile[wave][i][0];
#pragma unroll
      for (int j = 0; j < 2; ++j)
#pragma unroll
        for (int r = 0; r < 8; ++r)
          pt[(r + 8 * g) * 40 + j * 16 + ln] = (_Float16)sc[i][j][r];
    }

    // ---- O += P @ V_block (Vt gives contiguous B-fragments) ----
#pragma unroll
    for (int i = 0; i < 2; ++i) {
      const _Float16* pt = &ptile[wave][i][0];
      v16h pa = load_a_f16(pt + ln * 40 + 8 * g);   // A-layout read from LDS
#pragma unroll
      for (int ct = 0; ct < 4; ++ct) {
        v16h bv = load_b_f16(vp + (size_t)(ct * 16 + ln) * SEQ + kb + 8 * g);
        o[i][ct] = wmma_f16(pa, bv, o[i][ct]);
      }
    }
  }

  // ---- finalize: O / l, concat heads into Xh[B,S,DM] ----
#pragma unroll
  for (int i = 0; i < 2; ++i) {
    float inv[8];
#pragma unroll
    for (int r = 0; r < 8; ++r) inv[r] = 1.0f / lrow[i][r];
#pragma unroll
    for (int ct = 0; ct < 4; ++ct)
#pragma unroll
      for (int r = 0; r < 8; ++r) {
        const int s = qrow0 + i * 16 + r + 8 * g;
        const int d = ct * 16 + ln;
        xh[(((size_t)(b * SEQ + s)) << 10) + h * DK + d] =
            (_Float16)(o[i][ct][r] * inv[r]);
      }
  }
}

// ---------------------------------------------------------------------------
// Kernel 3: output projection.  out = Xh @ Wo + bo  (f32 result)
// ---------------------------------------------------------------------------
__global__ __launch_bounds__(128)
void out_proj_kernel(const _Float16* __restrict__ xh,
                     const _Float16* __restrict__ wto,
                     const float* __restrict__ bo, float* __restrict__ out) {
  const int lane = threadIdx.x & 31;
  const int wave = threadIdx.x >> 5;
  const int g    = lane >> 4;
  const int ln   = lane & 15;
  const int row0 = blockIdx.x * 128 + wave * 32;
  const int col0 = blockIdx.y * 64;

  v8f acc[2][4];
#pragma unroll
  for (int i = 0; i < 2; ++i)
#pragma unroll
    for (int j = 0; j < 4; ++j) acc[i][j] = (v8f)(0.0f);

  for (int kb = 0; kb < DM; kb += 32) {
    v16h a[2];
#pragma unroll
    for (int i = 0; i < 2; ++i)
      a[i] = load_a_f16(xh + (size_t)(row0 + i * 16 + ln) * DM + kb + 8 * g);
#pragma unroll
    for (int j = 0; j < 4; ++j) {
      v16h bf = load_b_f16(wto + (size_t)(col0 + j * 16 + ln) * DM + kb + 8 * g);
#pragma unroll
      for (int i = 0; i < 2; ++i) acc[i][j] = wmma_f16(a[i], bf, acc[i][j]);
    }
  }

#pragma unroll
  for (int j = 0; j < 4; ++j) {
    const int n = col0 + j * 16 + ln;
    const float bval = bo[n];
#pragma unroll
    for (int i = 0; i < 2; ++i)
#pragma unroll
      for (int r = 0; r < 8; ++r) {
        const int m = row0 + i * 16 + r + 8 * g;
        out[(size_t)m * DM + n] = acc[i][j][r] + bval;
      }
  }
}

// ---------------------------------------------------------------------------
// Host-side launch (graph-capture safe: stream-ordered kernels only).
// Workspace map (bytes):
//   [ 0, 8M)  : Wt_q/Wt_k/Wt_v/Wt_o   f16, 2 MB each
//   [ 8M,24M) : Qh  [B,H,S,DK]  f16
//   [24M,40M) : Kh  [B,H,S,DK]  f16
//   [40M,56M) : Vt  [B,H,DK,S]  f16
//   [56M,72M) : Xh  [B,S,DM]    f16
// ---------------------------------------------------------------------------
extern "C" void kernel_launch(void* const* d_in, const int* in_sizes, int n_in,
                              void* d_out, int out_size, void* d_ws,
                              size_t ws_size, hipStream_t stream) {
  (void)in_sizes; (void)n_in; (void)out_size; (void)ws_size;

  const float* q  = (const float*)d_in[0];
  const float* k  = (const float*)d_in[1];
  const float* v  = (const float*)d_in[2];
  const float* wq = (const float*)d_in[3];
  const float* bq = (const float*)d_in[4];
  const float* wk = (const float*)d_in[5];
  const float* bk = (const float*)d_in[6];
  const float* wv = (const float*)d_in[7];
  const float* bv = (const float*)d_in[8];
  const float* wo = (const float*)d_in[9];
  const float* bo = (const float*)d_in[10];

  char* ws = (char*)d_ws;
  const size_t MB = 1u << 20;
  _Float16* wtq = (_Float16*)(ws + 0 * MB);
  _Float16* wtk = (_Float16*)(ws + 2 * MB);
  _Float16* wtv = (_Float16*)(ws + 4 * MB);
  _Float16* wto = (_Float16*)(ws + 6 * MB);
  _Float16* qh  = (_Float16*)(ws + 8 * MB);
  _Float16* kh  = (_Float16*)(ws + 24 * MB);
  _Float16* vt  = (_Float16*)(ws + 40 * MB);
  _Float16* xh  = (_Float16*)(ws + 56 * MB);

  {
    dim3 blk(32, 8), grd(32, 32);
    wtrans_kernel<<<grd, blk, 0, stream>>>(wq, wtq);
    wtrans_kernel<<<grd, blk, 0, stream>>>(wk, wtk);
    wtrans_kernel<<<grd, blk, 0, stream>>>(wv, wtv);
    wtrans_kernel<<<grd, blk, 0, stream>>>(wo, wto);
  }

  qkv_proj_kernel<<<dim3(MROWS / 128, DM / 64, 3), 128, 0, stream>>>(
      q, k, v, wtq, wtk, wtv, bq, bk, bv, qh, kh, vt);

  attn_kernel<<<dim3(SEQ / 128, BATCH * NHEAD), 128, 0, stream>>>(qh, kh, vt,
                                                                  xh);

  out_proj_kernel<<<dim3(MROWS / 128, DM / 64), 128, 0, stream>>>(
      xh, wto, bo, (float*)d_out);
}
